// SpikingEmbeddingLayer_44650480009406
// MI455X (gfx1250) — compile-verified, matrix operand order checked by reading, not run on hardware
//
#include <hip/hip_runtime.h>

// ---------------- problem constants ----------------
#define T_    4
#define B_    8
#define C_    64
#define H_    120
#define W_    160
#define OC_   128
#define HW_   (H_ * W_)        // 19200
#define NIMG_ (T_ * B_)        // 32
#define KTOT_ 576              // C_ * 9
#define NKB_  18               // KTOT_ / 32

typedef __attribute__((ext_vector_type(16))) __bf16 v16bf;
typedef __attribute__((ext_vector_type(8)))  float  v8f;
typedef __attribute__((ext_vector_type(4)))  unsigned int u32x4;
typedef __attribute__((ext_vector_type(8)))  int i32x8;
typedef __attribute__((ext_vector_type(4)))  int i32x4;

union BF16Frag {
  v16bf v;
  unsigned short u[16];
  uint4 q[2];
};

__device__ __forceinline__ unsigned short f2bf(float f) {
  union { float f; unsigned u; } a;
  a.f = f;
  unsigned r = a.u + 0x7FFFu + ((a.u >> 16) & 1u);  // round-to-nearest-even
  return (unsigned short)(r >> 16);
}

// k-offset within a 32-wide K block for (lane, slot i): lanes 0-15 hold
// K {0..7,16..23}, lanes 16-31 hold K {8..15,24..31}.
__device__ __forceinline__ unsigned kmap(int lane, int i) {
  unsigned base = (lane & 16) ? 8u : 0u;
  return base + (unsigned)i + ((i & 8) ? 8u : 0u);
}

// ---------------------------------------------------------------
// TDM helper: 1-D bulk copy global -> LDS (ndwords 32-bit words).
// D# built per CDNA5 ISA 8.3/8.4: group0 = {flags, lds_addr, global_addr,
// type=2}; group1 = sizes/strides; groups 2/3 unused (<=2D).
// clang-23 toolchain: 6-arg builtin (extra int32x8 group before cpol).
// ---------------------------------------------------------------
__device__ __forceinline__ void tdm_load_1d(unsigned lds_off,
                                            const void* gaddr,
                                            unsigned ndwords) {
  unsigned long long ga = (unsigned long long)(uintptr_t)gaddr;
  u32x4 g0;
  g0[0] = 1u;                                          // count=1, user desc
  g0[1] = lds_off;                                     // lds_addr (bytes)
  g0[2] = (unsigned)(ga & 0xFFFFFFFFu);                // global_addr[31:0]
  g0[3] = (unsigned)((ga >> 32) & 0x1FFFFFFu) | (2u << 30); // addr[56:32] | type=2
  i32x8 g1;
  g1[0] = (int)(2u << 16);                             // data_size = 4 bytes
  g1[1] = (int)((ndwords & 0xFFFFu) << 16);            // tensor_dim0[15:0]
  g1[2] = (int)((ndwords >> 16) & 0xFFFFu) | (1 << 16); // tensor_dim0 hi | tensor_dim1=1
  g1[3] = (int)((ndwords & 0xFFFFu) << 16);            // tile_dim0 = ndwords
  g1[4] = 0;                                           // tile_dim1/2 unused
  g1[5] = (int)ndwords;                                // tensor_dim0_stride lo
  g1[6] = 0;
  g1[7] = 0;
  i32x4 z4 = {0, 0, 0, 0};
  i32x8 z8 = {0, 0, 0, 0, 0, 0, 0, 0};
  __builtin_amdgcn_tensor_load_to_lds(g0, g1, z4, z4, z8, 0);
}

// ---------------------------------------------------------------
// Kernel 1: pack conv weights into bf16 WMMA A-fragment layout.
// wfrag[octile(8)][kb(18)][lane(32)][slot(16)], k = tap*64 + c.
// ---------------------------------------------------------------
__global__ __launch_bounds__(256) void pack_w(const float* __restrict__ w,
                                              unsigned short* __restrict__ wfrag) {
  int t = blockIdx.x * 256 + threadIdx.x;      // 0 .. 73727
  int i    = t & 15;
  int lane = (t >> 4) & 31;
  int kt   = t >> 9;                           // 0 .. 143
  int kb   = kt % NKB_;
  int octile = kt / NKB_;                      // 0 .. 7
  int k  = kb * 32 + (int)kmap(lane, i);       // 0 .. 575
  int oc = octile * 16 + (lane & 15);
  int tap = k >> 6;                            // 0..8  (dy*3+dx)
  int c   = k & 63;
  int dy = tap / 3, dx = tap % 3;
  // conv_w layout [OC][C][3][3]
  wfrag[t] = f2bf(w[(size_t)oc * (C_ * 9) + (size_t)c * 9 + dy * 3 + dx]);
}

// ---------------------------------------------------------------
// Kernel 2: implicit-GEMM conv via v_wmma_f32_16x16x32_bf16.
// Workgroup = 256 thr (8 waves) -> tile 128 oc x (4 rows x 16 cols).
// LDS x-tile is stored c-INNERMOST so each B fragment = 2x ds_load_b128.
// kb loop fully unrolled: tap/dy/dx and all LDS offsets become immediates.
// ---------------------------------------------------------------
__global__ __launch_bounds__(256) void conv_wmma(const float* __restrict__ x,
                                                 const unsigned short* __restrict__ wfrag,
                                                 float* __restrict__ y) {
  __shared__ unsigned short lds_x[6][18][C_];   // [row][col][c] bf16, 13.8 KB

  const int tid = threadIdx.x;
  const int bx  = blockIdx.x;                   // 0 .. 9599
  const int n   = bx / 300;                     // image
  const int rem = bx % 300;
  const int h0  = (rem / 10) * 4;
  const int w0  = (rem % 10) * 16;

  // Stage input halo tile into LDS (f32 -> bf16), zero-padded borders.
  // Read order keeps global accesses w-contiguous; LDS store transposes to
  // c-innermost.
  for (int idx = tid; idx < C_ * 6 * 18; idx += 256) {
    int col = idx % 18;
    int r   = (idx / 18) % 6;
    int c   = idx / 108;
    int hs = h0 - 1 + r;
    int ws = w0 - 1 + col;
    float v = 0.0f;
    if (hs >= 0 && hs < H_ && ws >= 0 && ws < W_)
      v = x[(((size_t)n * C_ + c) * H_ + hs) * W_ + ws];
    lds_x[r][col][c] = f2bf(v);
  }
  __syncthreads();

  const int wave = tid >> 5;
  const int lane = tid & 31;
  const int og   = wave >> 1;        // 0..3: oc group of 32
  const int pw   = wave & 1;         // 0..1: row pair
  const int ln   = lane & 15;
  const unsigned base = (lane & 16) ? 8u : 0u;

  // Per-wave A-fragment base pointers (stride between kb steps is constant).
  const unsigned short* wf0 = wfrag + ((size_t)((og * 2 + 0) * NKB_ * 32 + lane) << 4);
  const unsigned short* wf1 = wfrag + ((size_t)((og * 2 + 1) * NKB_ * 32 + lane) << 4);

  v8f zero = {};
  v8f acc00 = zero, acc01 = zero, acc10 = zero, acc11 = zero;

#pragma unroll
  for (int kb = 0; kb < NKB_; ++kb) {
    const int tap = kb >> 1;                 // fixed 3x3 tap for this K block
    const int dy = tap / 3, dx = tap % 3;
    const unsigned chalf = (unsigned)(kb & 1) * 32u;

    // A fragments: contiguous 32B per lane from packed layout (L2-hot).
    BF16Frag a0, a1;
    a0.v = *(const v16bf*)(wf0 + ((size_t)(kb * 32) << 4));
    a1.v = *(const v16bf*)(wf1 + ((size_t)(kb * 32) << 4));

    // B fragments: two contiguous 8-channel runs -> 2x ds_load_b128 each.
    BF16Frag b0, b1;
    {
      const unsigned short* s = &lds_x[2 * pw + 0 + dy][ln + dx][chalf + base];
      b0.q[0] = *(const uint4*)s;            // c = chalf+base .. +7
      b0.q[1] = *(const uint4*)(s + 16);     // c = chalf+base+16 .. +23
    }
    {
      const unsigned short* s = &lds_x[2 * pw + 1 + dy][ln + dx][chalf + base];
      b1.q[0] = *(const uint4*)s;
      b1.q[1] = *(const uint4*)(s + 16);
    }

    acc00 = __builtin_amdgcn_wmma_f32_16x16x32_bf16(false, a0.v, false, b0.v,
                                                    (short)0, acc00, false, false);
    acc01 = __builtin_amdgcn_wmma_f32_16x16x32_bf16(false, a0.v, false, b1.v,
                                                    (short)0, acc01, false, false);
    acc10 = __builtin_amdgcn_wmma_f32_16x16x32_bf16(false, a1.v, false, b0.v,
                                                    (short)0, acc10, false, false);
    acc11 = __builtin_amdgcn_wmma_f32_16x16x32_bf16(false, a1.v, false, b1.v,
                                                    (short)0, acc11, false, false);
  }

  // Store D tiles: VGPR r -> M = r + (lane>=16 ? 8 : 0), N = lane&15.
  const int mhi = (lane & 16) ? 8 : 0;
  v8f accs[4] = {acc00, acc01, acc10, acc11};
#pragma unroll
  for (int oi = 0; oi < 2; ++oi) {
#pragma unroll
    for (int ri = 0; ri < 2; ++ri) {
      const v8f a = accs[oi * 2 + ri];
      const int hh = h0 + 2 * pw + ri;
#pragma unroll
      for (int r = 0; r < 8; ++r) {
        int oc = og * 32 + oi * 16 + r + mhi;
        size_t yi = (((size_t)n * OC_ + oc) * H_ + hh) * W_ + (w0 + ln);
        y[yi] = a[r];
      }
    }
  }
}

// ---------------------------------------------------------------
// Kernel 3: per-channel BN batch stats. Streams y through LDS with the
// Tensor Data Mover (double-buffered 4800-dword chunks), deterministic
// tree reduction, emits fused scale/shift.
// ---------------------------------------------------------------
#define CHUNK_ 4800                 // dwords per TDM transfer (HW_/4)
__global__ __launch_bounds__(256) void bn_stats(const float* __restrict__ y,
                                                const float* __restrict__ gamma,
                                                const float* __restrict__ beta,
                                                float* __restrict__ scale,
                                                float* __restrict__ shift) {
  __shared__ float lds_y[2][CHUNK_];          // 38.4 KB double buffer
  __shared__ float s1[256];
  __shared__ float s2[256];
  const int oc  = blockIdx.x;
  const int tid = threadIdx.x;
  const int NCH = NIMG_ * (HW_ / CHUNK_);     // 128 chunks

  float sum = 0.0f, sq = 0.0f;

  if (tid < 32) {
    tdm_load_1d((unsigned)(uintptr_t)(void*)&lds_y[0][0],
                y + ((size_t)(0 * OC_ + oc)) * HW_, CHUNK_);
  }
  for (int i = 0; i < NCH; ++i) {
    if (i + 1 < NCH) {
      if (tid < 32) {
        int n = (i + 1) >> 2, part = (i + 1) & 3;
        tdm_load_1d((unsigned)(uintptr_t)(void*)&lds_y[(i + 1) & 1][0],
                    y + ((size_t)(n * OC_ + oc)) * HW_ + (size_t)part * CHUNK_,
                    CHUNK_);
        __builtin_amdgcn_s_wait_tensorcnt(1);   // chunk i complete
      }
    } else {
      if (tid < 32) __builtin_amdgcn_s_wait_tensorcnt(0);
    }
    __syncthreads();
    const float* buf = lds_y[i & 1];
    for (int j = tid; j < CHUNK_; j += 256) {
      float v = buf[j];
      sum += v;
      sq  += v * v;
    }
    __syncthreads();
  }

  s1[tid] = sum; s2[tid] = sq;
  __syncthreads();
  for (int off = 128; off > 0; off >>= 1) {
    if (tid < off) { s1[tid] += s1[tid + off]; s2[tid] += s2[tid + off]; }
    __syncthreads();
  }
  if (tid == 0) {
    const float cnt = (float)NIMG_ * (float)HW_;
    float mean = s1[0] / cnt;
    float var  = s2[0] / cnt - mean * mean;
    float inv  = rsqrtf(var + 1e-5f) * gamma[oc];
    scale[oc] = inv;
    shift[oc] = beta[oc] - mean * inv;
  }
}

// ---------------------------------------------------------------
// Kernel 4: fused BN + 4-step LIF (hard reset, v_reset=0, tau=2).
// ---------------------------------------------------------------
__global__ __launch_bounds__(256) void bn_lif(const float* __restrict__ y,
                                              const float* __restrict__ scale,
                                              const float* __restrict__ shift,
                                              float* __restrict__ out) {
  size_t idx = (size_t)blockIdx.x * 256 + threadIdx.x;  // < 19,660,800
  int p  = (int)(idx % HW_);
  int oc = (int)((idx / HW_) % OC_);
  int b  = (int)(idx / ((size_t)HW_ * OC_));
  float sc = scale[oc];
  float sh = shift[oc];
  float v = 0.0f;
#pragma unroll
  for (int t = 0; t < T_; ++t) {
    size_t yi = ((size_t)(t * B_ + b) * OC_ + oc) * HW_ + p;
    float yv = y[yi] * sc + sh;
    float h  = v + (yv - v) * 0.5f;           // decay_input, tau=2
    bool  s  = (h - 1.0f) >= 0.0f;            // spike threshold 1.0
    out[yi] = s ? 1.0f : 0.0f;
    v = s ? 0.0f : h;                          // hard reset to 0
  }
}

// ---------------------------------------------------------------
extern "C" void kernel_launch(void* const* d_in, const int* in_sizes, int n_in,
                              void* d_out, int out_size, void* d_ws, size_t ws_size,
                              hipStream_t stream) {
  const float* x     = (const float*)d_in[0];  // [4,8,64,120,160]
  const float* w     = (const float*)d_in[1];  // [128,64,3,3]
  const float* gamma = (const float*)d_in[2];  // [128]
  const float* beta  = (const float*)d_in[3];  // [128]
  float* out = (float*)d_out;                  // [4,8,128,120,160]

  char* ws = (char*)d_ws;
  const size_t Y_BYTES  = (size_t)NIMG_ * OC_ * HW_ * sizeof(float);           // 314,572,800
  const size_t WF_BYTES = (size_t)8 * NKB_ * 32 * 16 * sizeof(unsigned short); // 147,456
  float* y_ws = (float*)ws;
  unsigned short* wfrag = (unsigned short*)(ws + Y_BYTES);
  float* scale = (float*)(ws + Y_BYTES + WF_BYTES);
  float* shift = scale + OC_;

  pack_w<<<288, 256, 0, stream>>>(w, wfrag);
  conv_wmma<<<NIMG_ * (H_ / 4) * (W_ / 16), 256, 0, stream>>>(x, wfrag, y_ws);
  bn_stats<<<OC_, 256, 0, stream>>>(y_ws, gamma, beta, scale, shift);
  bn_lif<<<(B_ * OC_ * HW_) / 256, 256, 0, stream>>>(y_ws, scale, shift, out);
}
